// MultiBandGNN_48034914238741
// MI455X (gfx1250) — matrix operand order repaired and checked by reading, not compiled, not run on hardware
//
#include <hip/hip_runtime.h>
#include <hip/hip_bf16.h>

// ---------------- problem constants (match reference) ----------------
static const int NN  = 50000;    // nodes
static const int NE  = 800000;   // edges (before self loops)
static const int IND = 128;      // input dim
static const int NG  = 100;      // graphs
static const int HEADS = 4, HDIM = 32;

// ---------------- WMMA types ----------------
typedef __attribute__((ext_vector_type(16))) __bf16 v16bf;
typedef __attribute__((ext_vector_type(8)))  __bf16 v8bf;   // one b128 load
typedef __attribute__((ext_vector_type(8)))  float  v8f;

__device__ __forceinline__ __bf16 f2bf(float f) {
  // round-to-nearest-even f32 -> bf16
  unsigned u = __float_as_uint(f);
  unsigned r = u + 0x7FFFu + ((u >> 16) & 1u);
  unsigned short h = (unsigned short)(r >> 16);
  return __builtin_bit_cast(__bf16, h);
}

#define GSTRIDE(i, total)                                                     \
  for (long long i = (long long)blockIdx.x * blockDim.x + threadIdx.x;        \
       i < (total); i += (long long)gridDim.x * blockDim.x)

// ---------------- f32 -> bf16 pre-pack (pair-packed stores) ----------------
__global__ void cvt_f32_bf16(const float* __restrict__ in, __bf16* __restrict__ out,
                             long long n) {           // n even
  GSTRIDE(i, n >> 1) {
    long long b = i << 1;
    unsigned short lo = __builtin_bit_cast(unsigned short, f2bf(in[b]));
    unsigned short hi = __builtin_bit_cast(unsigned short, f2bf(in[b + 1]));
    ((unsigned*)out)[i] = (unsigned)lo | ((unsigned)hi << 16);
  }
}

// ---------------- GEMM: C[M,N] = A[M,K] @ B[K,N], bf16 in / f32 out --------
// One wave per 16x16 C tile; M,N multiples of 16; K multiple of 32.
// A fragment per lane: K = [ko+8*hi, +8) and [ko+16+8*hi, +8)  (2x b128)
// B fragment per lane: col N = nt*16+m, K = [ko+8*hi, +8) and [ko+16+8*hi, +8)
__global__ void __launch_bounds__(256)
gemm_bf16_wmma(const __bf16* __restrict__ A, const __bf16* __restrict__ B,
               float* __restrict__ C, int M, int K, int N) {
  int wave = (int)(((long long)blockIdx.x * blockDim.x + threadIdx.x) >> 5);
  int lane = threadIdx.x & 31;            // branch below is wave-uniform
  int ntiles = N >> 4;
  int mtiles = M >> 4;
  if (wave >= mtiles * ntiles) return;    // EXEC stays all-1s for live waves
  int mt = wave / ntiles;
  int nt = wave - mt * ntiles;
  int m  = lane & 15;
  int hi = lane >> 4;

  const __bf16* arow = A + (size_t)(mt * 16 + m) * K;
  v8f acc = {};
  for (int ko = 0; ko < K; ko += 32) {
    v8bf alo = *(const v8bf*)(arow + ko + 8 * hi);
    v8bf ahi = *(const v8bf*)(arow + ko + 16 + 8 * hi);
    v16bf a = __builtin_shufflevector(alo, ahi, 0, 1, 2, 3, 4, 5, 6, 7,
                                      8, 9, 10, 11, 12, 13, 14, 15);
    const __bf16* bcol = B + (size_t)(ko + 8 * hi) * N + nt * 16 + m;
    v16bf b;
#pragma unroll
    for (int s = 0; s < 8; ++s) {
      b[s] = bcol[(size_t)s * N];
      b[8 + s] = bcol[(size_t)(16 + s) * N];
    }
    acc = __builtin_amdgcn_wmma_f32_16x16x32_bf16(
        false, a, false, b, (short)0, acc, false, false);
  }
  // C f32 layout: VGPR r -> row mt*16 + r + 8*hi, col nt*16 + m
  float* crow = C + (size_t)(mt * 16 + 8 * hi) * N + nt * 16 + m;
#pragma unroll
  for (int r = 0; r < 8; ++r) crow[(size_t)r * N] = acc[r];
}

// ---------------- degree / norm ----------------
__global__ void set_ones(float* p, long long n) { GSTRIDE(i, n) p[i] = 1.0f; }

__global__ void deg_count(const int* __restrict__ dst, float* __restrict__ deg,
                          long long ne) {
  GSTRIDE(i, ne) atomicAdd(&deg[dst[i]], 1.0f);
}

__global__ void finish_dinv(float* p, long long n) {
  GSTRIDE(i, n) p[i] = rsqrtf(p[i]);
}

// ---------------- GCN scatter: agg[dst] += dinv[s]*dinv[d]*xw[src] --------
__global__ void gcn_scatter(const float* __restrict__ xw, const float* __restrict__ dinv,
                            const int* __restrict__ src, const int* __restrict__ dst,
                            float* __restrict__ agg, int F, long long nreal,
                            long long nnodes) {
  long long total = (nreal + nnodes) * F;
  GSTRIDE(i, total) {
    long long e = i / F;
    int f = (int)(i - e * F);
    int s, d;
    if (e < nreal) { s = src[e]; d = dst[e]; }
    else           { s = d = (int)(e - nreal); }   // self loop
    float nrm = dinv[s] * dinv[d];
    atomicAdd(&agg[(size_t)d * F + f], nrm * xw[(size_t)s * F + f]);
  }
}

// fused ReLU(+bias) epilogue emitting bf16 (next layer's GEMM operand)
__global__ void relu_bias_bf16(const float* __restrict__ agg, const float* __restrict__ b,
                               __bf16* __restrict__ out, int F, long long nnodes) {
  GSTRIDE(i, (nnodes * F) >> 1) {          // F even
    long long base = i << 1;
    int f = (int)(base % F);
    float v0 = fmaxf(agg[base]     + b[f],     0.0f);
    float v1 = fmaxf(agg[base + 1] + b[f + 1], 0.0f);
    unsigned short lo = __builtin_bit_cast(unsigned short, f2bf(v0));
    unsigned short hi = __builtin_bit_cast(unsigned short, f2bf(v1));
    ((unsigned*)out)[i] = (unsigned)lo | ((unsigned)hi << 16);
  }
}

// f32 ReLU(+bias) (GAT output path, stays f32)
__global__ void relu_bias_f32(float* __restrict__ h, const float* __restrict__ b,
                              int F, long long nnodes) {
  GSTRIDE(i, nnodes * F) {
    float v = h[i] + b[i % F];
    h[i] = v > 0.0f ? v : 0.0f;
  }
}

// ---------------- GAT ----------------
__global__ void att_scores(const float* __restrict__ hg,
                           const float* __restrict__ att_src,
                           const float* __restrict__ att_dst,
                           float* __restrict__ a_s, float* __restrict__ a_d,
                           long long nnodes) {
  GSTRIDE(i, nnodes * HEADS) {
    long long n = i >> 2;
    int h = (int)(i & 3);
    const float* v = hg + (size_t)n * (HEADS * HDIM) + h * HDIM;
    float ss = 0.f, sd = 0.f;
#pragma unroll 8
    for (int d = 0; d < HDIM; ++d) {
      ss += v[d] * att_src[h * HDIM + d];
      sd += v[d] * att_dst[h * HDIM + d];
    }
    a_s[i] = ss; a_d[i] = sd;
  }
}

__global__ void init_neg_inf(float* p, long long n) {
  GSTRIDE(i, n) p[i] = -__builtin_inff();
}

__device__ __forceinline__ void atomicMaxF32(float* addr, float val) {
  if (val >= 0.f) atomicMax((int*)addr, __float_as_int(val));
  else            atomicMin((unsigned int*)addr, __float_as_uint(val));
}

__device__ __forceinline__ float leaky02(float v) {
  return v > 0.f ? v : 0.2f * v;
}

__global__ void edge_max(const float* __restrict__ a_s, const float* __restrict__ a_d,
                         const int* __restrict__ src, const int* __restrict__ dst,
                         float* __restrict__ mmax, long long nreal, long long nnodes) {
  GSTRIDE(i, (nreal + nnodes) * HEADS) {
    long long e = i >> 2;
    int h = (int)(i & 3);
    int s, d;
    if (e < nreal) { s = src[e]; d = dst[e]; } else { s = d = (int)(e - nreal); }
    float v = leaky02(a_s[(size_t)s * HEADS + h] + a_d[(size_t)d * HEADS + h]);
    atomicMaxF32(&mmax[(size_t)d * HEADS + h], v);
  }
}

__global__ void edge_expsum(const float* __restrict__ a_s, const float* __restrict__ a_d,
                            const int* __restrict__ src, const int* __restrict__ dst,
                            const float* __restrict__ mmax, float* __restrict__ ssum,
                            long long nreal, long long nnodes) {
  GSTRIDE(i, (nreal + nnodes) * HEADS) {
    long long e = i >> 2;
    int h = (int)(i & 3);
    int s, d;
    if (e < nreal) { s = src[e]; d = dst[e]; } else { s = d = (int)(e - nreal); }
    float v = leaky02(a_s[(size_t)s * HEADS + h] + a_d[(size_t)d * HEADS + h]);
    atomicAdd(&ssum[(size_t)d * HEADS + h], expf(v - mmax[(size_t)d * HEADS + h]));
  }
}

__global__ void gat_scatter(const float* __restrict__ hg,
                            const float* __restrict__ a_s, const float* __restrict__ a_d,
                            const float* __restrict__ mmax, const float* __restrict__ ssum,
                            const int* __restrict__ src, const int* __restrict__ dst,
                            float* __restrict__ out3, long long nreal, long long nnodes) {
  const int FW = HEADS * HDIM; // 128
  GSTRIDE(i, (nreal + nnodes) * FW) {
    long long e = i / FW;
    int r = (int)(i - e * FW);
    int h = r >> 5;
    int s, d;
    if (e < nreal) { s = src[e]; d = dst[e]; } else { s = d = (int)(e - nreal); }
    float v = leaky02(a_s[(size_t)s * HEADS + h] + a_d[(size_t)d * HEADS + h]);
    float alpha = expf(v - mmax[(size_t)d * HEADS + h]) / ssum[(size_t)d * HEADS + h];
    atomicAdd(&out3[(size_t)d * FW + r], alpha * hg[(size_t)s * FW + r]);
  }
}

__global__ void head_mean_relu(const float* __restrict__ out3,
                               const float* __restrict__ bg,
                               float* __restrict__ h3, long long nnodes) {
  GSTRIDE(i, nnodes * HDIM) {
    long long n = i >> 5;
    int d = (int)(i & 31);
    const float* o = out3 + (size_t)n * (HEADS * HDIM) + d;
    float v = 0.25f * (o[0] + o[HDIM] + o[2 * HDIM] + o[3 * HDIM]) + bg[d];
    h3[i] = v > 0.f ? v : 0.f;
  }
}

__global__ void pool_sum(const float* __restrict__ h3, const int* __restrict__ batch,
                         float* __restrict__ pooled, long long nnodes) {
  GSTRIDE(i, nnodes * HDIM) {
    long long n = i >> 5;
    int d = (int)(i & 31);
    atomicAdd(&pooled[(size_t)batch[n] * HDIM + d], h3[i]);
  }
}

__global__ void fc_out(const float* __restrict__ pooled, const float* __restrict__ Wfc,
                       const float* __restrict__ bfc, float* __restrict__ out) {
  int i = blockIdx.x * blockDim.x + threadIdx.x;
  if (i >= NG * 3) return;
  int g = i / 3, c = i - g * 3;
  float acc = bfc[c];
#pragma unroll 8
  for (int d = 0; d < HDIM; ++d) acc += pooled[g * HDIM + d] * Wfc[d * 3 + c];
  out[i] = acc;
}

// ---------------- host side ----------------
static inline int nblk(long long total, int bs = 256) {
  long long b = (total + bs - 1) / bs;
  if (b > (1LL << 20)) b = (1LL << 20);
  if (b < 1) b = 1;
  return (int)b;
}

extern "C" void kernel_launch(void* const* d_in, const int* in_sizes, int n_in,
                              void* d_out, int out_size, void* d_ws, size_t ws_size,
                              hipStream_t stream) {
  (void)in_sizes; (void)n_in; (void)out_size; (void)ws_size;
  const float* x       = (const float*)d_in[0];
  const int*   eidx    = (const int*)d_in[1];   // [2, NE]
  const int*   batch   = (const int*)d_in[2];
  const float* W1      = (const float*)d_in[3];
  const float* b1      = (const float*)d_in[4];
  const float* W2      = (const float*)d_in[5];
  const float* b2      = (const float*)d_in[6];
  const float* Wg      = (const float*)d_in[7];
  const float* att_src = (const float*)d_in[8];
  const float* att_dst = (const float*)d_in[9];
  const float* bg      = (const float*)d_in[10];
  const float* Wfc     = (const float*)d_in[11];
  const float* bfc     = (const float*)d_in[12];
  float* out = (float*)d_out;

  const int* src = eidx;
  const int* dst = eidx + NE;

  // scratch carving (256B aligned)
  char* base = (char*)d_ws;
  size_t off = 0;
  auto carve = [&](size_t bytes) {
    char* p = base + off;
    off += (bytes + 255) & ~(size_t)255;
    return p;
  };
  float*  dinv   = (float*)carve((size_t)NN * 4);
  __bf16* xbf    = (__bf16*)carve((size_t)NN * IND * 2);
  __bf16* W1bf   = (__bf16*)carve((size_t)IND * 64 * 2);
  __bf16* W2bf   = (__bf16*)carve((size_t)64 * 32 * 2);
  __bf16* Wgbf   = (__bf16*)carve((size_t)32 * 128 * 2);
  float*  xw1    = (float*)carve((size_t)NN * 64 * 4);
  float*  agg1   = (float*)carve((size_t)NN * 64 * 4);
  __bf16* h1bf   = (__bf16*)carve((size_t)NN * 64 * 2);
  float*  xw2    = (float*)carve((size_t)NN * 32 * 4);
  float*  agg2   = (float*)carve((size_t)NN * 32 * 4);
  __bf16* h2bf   = (__bf16*)carve((size_t)NN * 32 * 2);
  float*  hg     = (float*)carve((size_t)NN * 128 * 4);
  float*  a_s    = (float*)carve((size_t)NN * HEADS * 4);
  float*  a_d    = (float*)carve((size_t)NN * HEADS * 4);
  float*  mmax   = (float*)carve((size_t)NN * HEADS * 4);
  float*  ssum   = (float*)carve((size_t)NN * HEADS * 4);
  float*  out3   = (float*)carve((size_t)NN * 128 * 4);
  float*  h3     = (float*)carve((size_t)NN * 32 * 4);
  float*  pooled = (float*)carve((size_t)NG * HDIM * 4);

  const long long nn = NN, ne = NE;

  // ---- degree / symmetric norm (self loops give deg >= 1) ----
  set_ones<<<nblk(nn), 256, 0, stream>>>(dinv, nn);
  deg_count<<<nblk(ne), 256, 0, stream>>>(dst, dinv, ne);
  finish_dinv<<<nblk(nn), 256, 0, stream>>>(dinv, nn);

  // ---- pre-pack bf16 operands ----
  cvt_f32_bf16<<<nblk((long long)NN * IND / 2), 256, 0, stream>>>(x, xbf, (long long)NN * IND);
  cvt_f32_bf16<<<nblk(IND * 64 / 2), 256, 0, stream>>>(W1, W1bf, IND * 64);
  cvt_f32_bf16<<<nblk(64 * 32 / 2), 256, 0, stream>>>(W2, W2bf, 64 * 32);
  cvt_f32_bf16<<<nblk(32 * 128 / 2), 256, 0, stream>>>(Wg, Wgbf, 32 * 128);

  // ---- GCN layer 1: xw1 = x @ W1 ; scatter ; relu(+b1) -> bf16 ----
  {
    long long waves = (long long)(NN / 16) * (64 / 16);
    gemm_bf16_wmma<<<nblk(waves * 32), 256, 0, stream>>>(xbf, W1bf, xw1, NN, IND, 64);
  }
  hipMemsetAsync(agg1, 0, (size_t)NN * 64 * 4, stream);
  gcn_scatter<<<nblk((ne + nn) * 64), 256, 0, stream>>>(xw1, dinv, src, dst, agg1, 64, ne, nn);
  relu_bias_bf16<<<nblk(nn * 64 / 2), 256, 0, stream>>>(agg1, b1, h1bf, 64, nn);

  // ---- GCN layer 2: xw2 = h1 @ W2 ; scatter ; relu(+b2) -> bf16 ----
  {
    long long waves = (long long)(NN / 16) * (32 / 16);
    gemm_bf16_wmma<<<nblk(waves * 32), 256, 0, stream>>>(h1bf, W2bf, xw2, NN, 64, 32);
  }
  hipMemsetAsync(agg2, 0, (size_t)NN * 32 * 4, stream);
  gcn_scatter<<<nblk((ne + nn) * 32), 256, 0, stream>>>(xw2, dinv, src, dst, agg2, 32, ne, nn);
  relu_bias_bf16<<<nblk(nn * 32 / 2), 256, 0, stream>>>(agg2, b2, h2bf, 32, nn);

  // ---- GAT: hg = h2 @ Wg ; per-dst softmax attention ; mean heads ----
  {
    long long waves = (long long)(NN / 16) * (128 / 16);
    gemm_bf16_wmma<<<nblk(waves * 32), 256, 0, stream>>>(h2bf, Wgbf, hg, NN, 32, 128);
  }
  att_scores<<<nblk(nn * HEADS), 256, 0, stream>>>(hg, att_src, att_dst, a_s, a_d, nn);
  init_neg_inf<<<nblk(nn * HEADS), 256, 0, stream>>>(mmax, nn * HEADS);
  hipMemsetAsync(ssum, 0, (size_t)NN * HEADS * 4, stream);
  edge_max<<<nblk((ne + nn) * HEADS), 256, 0, stream>>>(a_s, a_d, src, dst, mmax, ne, nn);
  edge_expsum<<<nblk((ne + nn) * HEADS), 256, 0, stream>>>(a_s, a_d, src, dst, mmax, ssum, ne, nn);
  hipMemsetAsync(out3, 0, (size_t)NN * 128 * 4, stream);
  gat_scatter<<<nblk((ne + nn) * 128), 256, 0, stream>>>(hg, a_s, a_d, mmax, ssum,
                                                         src, dst, out3, ne, nn);
  head_mean_relu<<<nblk(nn * HDIM), 256, 0, stream>>>(out3, bg, h3, nn);

  // ---- pooling + FC ----
  hipMemsetAsync(pooled, 0, (size_t)NG * HDIM * 4, stream);
  pool_sum<<<nblk(nn * HDIM), 256, 0, stream>>>(h3, batch, pooled, nn);
  fc_out<<<nblk(NG * 3), 256, 0, stream>>>(pooled, Wfc, bfc, out);
}
